// refine_attention_78752520339658
// MI455X (gfx1250) — matrix operand order repaired and checked by reading, not compiled, not run on hardware
//
#include <hip/hip_runtime.h>

#define BB 8
#define NN 2048
#define CC 256

typedef __attribute__((ext_vector_type(16))) _Float16 v16h;
typedef __attribute__((ext_vector_type(8)))  _Float16 v8h;
typedef __attribute__((ext_vector_type(8)))  float    v8f;
typedef __attribute__((ext_vector_type(4)))  float    v4f;

// ---------------------------------------------------------------------------
// Kernel 1: Vt[b, d, n] = (feat[b, n, :] . v_w[d, :]) + v_b[d], stored f16,
// TRANSPOSED (d-major) so the attention kernel's B-fragments are contiguous.
// One wave computes one 16(n) x 16(d) tile with 8x V_WMMA_F32_16X16X32_F16.
// ---------------------------------------------------------------------------
__global__ __launch_bounds__(128) void vproj_f16(const float* __restrict__ feat,
                                                 const float* __restrict__ v_w,
                                                 const float* __restrict__ v_b,
                                                 _Float16* __restrict__ Vt) {
  const int lane = threadIdx.x & 31;
  const int wave = threadIdx.x >> 5;
  int bid = blockIdx.x;
  const int dgroup = bid & 3;  bid >>= 2;   // C/64 = 4
  const int ntile  = bid & 127; bid >>= 7;  // N/16 = 128
  const int b      = bid;                   // B = 8
  const int n0 = ntile * 16;
  const int d0 = dgroup * 64 + wave * 16;
  const int half = lane >> 4;   // which half-wave
  const int l16  = lane & 15;

  // A fragment rows: M = l16 ; K layout: half 0 -> {0..7,16..23}, half 1 -> {8..15,24..31}
  const float* fA = feat + ((size_t)(b * NN + n0 + l16)) * CC;
  // B fragment: col N = l16 (row d0+l16 of v_w), K layout same as A: half 0 -> {0..7,16..23}, half 1 -> {8..15,24..31}
  const float* wB = v_w + (size_t)(d0 + l16) * CC + half * 8;

  v8f acc = {};
  #pragma unroll
  for (int k0 = 0; k0 < CC; k0 += 32) {
    const float* pa = fA + k0 + half * 8;
    v4f a0 = *(const v4f*)(pa);
    v4f a1 = *(const v4f*)(pa + 4);
    v4f a2 = *(const v4f*)(pa + 16);
    v4f a3 = *(const v4f*)(pa + 20);
    v16h a;
    #pragma unroll
    for (int i = 0; i < 4; ++i) {
      a[i]      = (_Float16)a0[i];
      a[4 + i]  = (_Float16)a1[i];
      a[8 + i]  = (_Float16)a2[i];
      a[12 + i] = (_Float16)a3[i];
    }
    const float* pb = wB + k0;
    v4f b0 = *(const v4f*)(pb);
    v4f b1 = *(const v4f*)(pb + 4);
    v4f b2 = *(const v4f*)(pb + 16);
    v4f b3 = *(const v4f*)(pb + 20);
    v16h bf;
    #pragma unroll
    for (int i = 0; i < 4; ++i) {
      bf[i]      = (_Float16)b0[i];
      bf[4 + i]  = (_Float16)b1[i];
      bf[8 + i]  = (_Float16)b2[i];
      bf[12 + i] = (_Float16)b3[i];
    }
    acc = __builtin_amdgcn_wmma_f32_16x16x32_f16(false, a, false, bf,
                                                 (short)0, acc, false, false);
  }

  const float bias = v_b[d0 + l16];             // D col is fixed per lane
  v8h o;
  #pragma unroll
  for (int v = 0; v < 8; ++v) o[v] = (_Float16)(acc[v] + bias);
  // D element (lane, vgpr v): row n = v + 8*half, col d = l16  -> transposed store,
  // 8 consecutive n values => one 16B store per lane.
  _Float16* dst = Vt + ((size_t)(b * CC + d0 + l16)) * NN + n0 + half * 8;
  *(v8h*)dst = o;
}

// ---------------------------------------------------------------------------
// Kernel 2: fused masked softmax + attn @ V (flash-style, single pass over corr).
// Block = 4 waves: (wave>>1) selects 16-row subtile, (wave&1) selects 128-col half.
// Each wave: 16 rows x 128 cols of output, 8 v8f accumulators.
// Software-pipelined: corr/mask double-buffered one chunk ahead; Vt fragments
// issued before the softmax VALU stream so L2 latency hides under exp2s.
// ---------------------------------------------------------------------------
__global__ __launch_bounds__(128) void attn_fused(const float* __restrict__ corr,
                                                  const float* __restrict__ mask,
                                                  const _Float16* __restrict__ Vt,
                                                  float* __restrict__ out) {
  const int lane = threadIdx.x & 31;
  const int wave = threadIdx.x >> 5;
  const int rsub = wave >> 1;
  const int csub = wave & 1;
  int bid = blockIdx.x;
  const int ntile = bid & 63; bid >>= 6;   // N/32 = 64 blocks of 32 rows
  const int b = bid;
  const int n0 = ntile * 32 + rsub * 16;
  const int d0 = csub * 128;
  const int half = lane >> 4;
  const int l16  = lane & 15;

  const float*    cR = corr + ((size_t)(b * NN + n0 + l16)) * NN + half * 8;
  const float*    mR = mask + (size_t)(n0 + l16) * NN + half * 8;
  const _Float16* vB = Vt + ((size_t)(b * CC + d0 + l16)) * NN + half * 8;

  v8f acc[8];
  #pragma unroll
  for (int j = 0; j < 8; ++j) acc[j] = (v8f){};
  float m = -1.0e30f, l = 0.0f;
  const float LOG2E = 1.4426950408889634f;

  // ---- prologue: buffer chunk 0 of corr/mask (A-fragment lane layout) ----
  v4f c0 = *(const v4f*)(cR);
  v4f c1 = *(const v4f*)(cR + 4);
  v4f c2 = *(const v4f*)(cR + 16);
  v4f c3 = *(const v4f*)(cR + 20);
  v4f q0 = *(const v4f*)(mR);
  v4f q1 = *(const v4f*)(mR + 4);
  v4f q2 = *(const v4f*)(mR + 16);
  v4f q3 = *(const v4f*)(mR + 20);

  for (int k0 = 0; k0 < NN; k0 += 32) {
    // ---- issue next chunk's score loads first (dead-end reload on last iter) ----
    const int kn = (k0 + 32 < NN) ? (k0 + 32) : k0;
    const float* cp = cR + kn;
    const float* mp = mR + kn;
    __builtin_prefetch(cp + 256, 0, 3);                 // corr, 8 chunks ahead
    v4f nc0 = *(const v4f*)(cp);
    v4f nc1 = *(const v4f*)(cp + 4);
    v4f nc2 = *(const v4f*)(cp + 16);
    v4f nc3 = *(const v4f*)(cp + 20);
    v4f nq0 = *(const v4f*)(mp);
    v4f nq1 = *(const v4f*)(mp + 4);
    v4f nq2 = *(const v4f*)(mp + 16);
    v4f nq3 = *(const v4f*)(mp + 20);

    // ---- issue all Vt B-fragments for the current chunk (L2-resident) ----
    const _Float16* vp = vB + k0;
    v16h bf[8];
    #pragma unroll
    for (int j = 0; j < 8; ++j) {
      v8h lo = *(const v8h*)(vp + (size_t)j * 16 * NN);
      v8h hi = *(const v8h*)(vp + (size_t)j * 16 * NN + 16);
      bf[j] = __builtin_shufflevector(lo, hi, 0, 1, 2, 3, 4, 5, 6, 7, 8, 9, 10, 11, 12, 13, 14, 15);
    }

    // ---- online softmax on buffered chunk k0 ----
    float s[16];
    #pragma unroll
    for (int i = 0; i < 4; ++i) {
      s[i]      = (c0[i] + q0[i]) * LOG2E;
      s[4 + i]  = (c1[i] + q1[i]) * LOG2E;
      s[8 + i]  = (c2[i] + q2[i]) * LOG2E;
      s[12 + i] = (c3[i] + q3[i]) * LOG2E;
    }
    // row max: local 16 + cross half-wave (lane r and r+16 hold halves of row r)
    float cmax = s[0];
    #pragma unroll
    for (int i = 1; i < 16; ++i) cmax = fmaxf(cmax, s[i]);
    cmax = fmaxf(cmax, __shfl_xor(cmax, 16, 32));
    const float mnew  = fmaxf(m, cmax);
    const float scale = exp2f(m - mnew);
    m = mnew;

    v16h p;
    float psum = 0.0f;
    #pragma unroll
    for (int i = 0; i < 16; ++i) {
      float e = exp2f(s[i] - mnew);
      psum += e;
      p[i] = (_Float16)e;
    }
    psum += __shfl_xor(psum, 16, 32);
    l = l * scale + psum;

    // Lazy accumulator rescale: only when any row's max actually moved.
    if (__ballot(scale < 1.0f)) {
      #pragma unroll
      for (int v = 0; v < 8; ++v) {
        float sc = __shfl(scale, v + half * 8, 32);   // scale for row v + 8*half
        #pragma unroll
        for (int j = 0; j < 8; ++j) acc[j][v] *= sc;
      }
    }

    // ---- P @ V for this chunk ----
    #pragma unroll
    for (int j = 0; j < 8; ++j)
      acc[j] = __builtin_amdgcn_wmma_f32_16x16x32_f16(false, p, false, bf[j],
                                                      (short)0, acc[j], false, false);

    // ---- rotate score double-buffer ----
    c0 = nc0; c1 = nc1; c2 = nc2; c3 = nc3;
    q0 = nq0; q1 = nq1; q2 = nq2; q3 = nq3;
  }

  const float linv = 1.0f / l;
  #pragma unroll
  for (int v = 0; v < 8; ++v) {
    const float li = __shfl(linv, v + half * 8, 32);
    float* op = out + ((size_t)(b * NN + n0 + half * 8 + v)) * CC + d0 + l16;
    #pragma unroll
    for (int j = 0; j < 8; ++j) op[j * 16] = acc[j][v] * li;
  }
}

// ---------------------------------------------------------------------------
extern "C" void kernel_launch(void* const* d_in, const int* in_sizes, int n_in,
                              void* d_out, int out_size, void* d_ws, size_t ws_size,
                              hipStream_t stream) {
  (void)in_sizes; (void)n_in; (void)out_size; (void)ws_size;
  const float* corr = (const float*)d_in[0];
  const float* feat = (const float*)d_in[1];
  /* d_in[2] histogram: unused by the reference computation */
  const float* mask = (const float*)d_in[3];
  const float* v_w  = (const float*)d_in[4];
  const float* v_b  = (const float*)d_in[5];
  float* out = (float*)d_out;
  _Float16* Vt = (_Float16*)d_ws;  // needs B*C*N*2 = 8 MB of workspace

  vproj_f16<<<dim3(BB * (NN / 16) * (CC / 64)), dim3(128), 0, stream>>>(feat, v_w, v_b, Vt);
  attn_fused<<<dim3(BB * (NN / 32)), dim3(128), 0, stream>>>(corr, mask, Vt, out);
}
